// ZoeDepthAttractorLayerUnnormed_20890720928634
// MI455X (gfx1250) — matrix operand order repaired and checked by reading, not compiled, not run on hardware
//
#include <hip/hip_runtime.h>
#include <math.h>

// MI455X / gfx1250, wave32.
// Matrix path : v_wmma_f32_16x16x32_f16 (f32 accumulate)
// Data path   : TENSOR_LOAD_TO_LDS (TDM DMA) double-buffered per k-step,
//               synchronized with s_wait_tensorcnt.

typedef _Float16 v16h __attribute__((ext_vector_type(16)));
typedef _Float16 v8h  __attribute__((ext_vector_type(8)));
typedef float    v8f  __attribute__((ext_vector_type(8)));
typedef unsigned int v4u __attribute__((ext_vector_type(4)));
typedef int          v4i __attribute__((ext_vector_type(4)));
typedef int          v8i __attribute__((ext_vector_type(8)));

namespace {
constexpr int kB    = 8;
constexpr int kCin  = 256;
constexpr int kCmid = 128;
constexpr int kAtt  = 16;
constexpr int kH    = 128, kW = 128;
constexpr int kHW   = kH * kW;          // 16384
constexpr int kPix  = kB * kHW;         // 131072
constexpr int kBins = 64, kPH = 64, kPW = 64;
constexpr int kOut1 = kB * kBins * kH * kW;  // 8388608 (one tuple element)

// Padded f16 LDS strides (16B aligned rows for ds_load_b128 fragments)
constexpr int W1S = 264;   // w1 row: 256 + 8
constexpr int W2S = 136;   // w2 row: 128 + 8
constexpr int HS  = 136;   // h  row: 128 + 8

// TDM-staged x tile: 32 channels x 16 pixels, padded 16 -> 20 dwords per row
// (TDM pad_interval = 16 dwords, pad_amount = 4 dwords) => conflict-free banks.
constexpr int XROW  = 20;                 // dwords per channel row in LDS
constexpr int XBUF  = 32 * XROW;          // 640 dwords = 2560 B per k-step tile
}

__device__ __forceinline__ v16h ld16h(const _Float16* p) {
  v8h lo = *(const v8h*)p;
  v8h hi = *(const v8h*)(p + 16);
  return __builtin_shufflevector(lo, hi, 0,1,2,3,4,5,6,7,8,9,10,11,12,13,14,15);
}

// Issue one TDM load: 16x32 f32 tile of x (pixel-contiguous dim0, channel dim1
// with stride HW) -> padded LDS tile at byte offset ldsOff.
// Descriptor per CDNA5 ISA (D# groups 0/1; groups 2/3 unused).
__device__ __forceinline__ void tdm_load_tile(const float* gptr, unsigned ldsOff) {
  const unsigned long long ga = (unsigned long long)(uintptr_t)gptr;
  v4u g0;
  g0.x = 1u;                                  // count=1, user-mode, no gather
  g0.y = ldsOff;                              // lds_addr (bytes)
  g0.z = (unsigned)ga;                        // global_addr[31:0]
  g0.w = (unsigned)(ga >> 32) | (2u << 30);   // global_addr[56:32] | type=2
  v8i g1;
  g1[0] = (int)((2u << 16)        // data_size = 2 (4-byte elements)
              | (1u << 20)        // pad_enable
              | (3u << 22)        // pad_interval: 16 dwords
              | (3u << 25));      // pad_amount : 4 dwords
  g1[1] = (int)((unsigned)kHW << 16);   // tensor_dim0[15:0] = 16384
  g1[2] = (int)(256u << 16);            // tensor_dim0 hi = 0, tensor_dim1 = 256
  g1[3] = (int)(16u << 16);             // tensor_dim1 hi = 0, tile_dim0 = 16
  g1[4] = 32;                           // tile_dim1 = 32, tile_dim2 = 0
  g1[5] = kHW;                          // tensor_dim0_stride[31:0] = 16384
  g1[6] = 0;                            // stride hi = 0, tensor_dim1_stride = 0
  g1[7] = 0;
  const v4i z4 = {0, 0, 0, 0};
#if __clang_major__ >= 23
  const v8i z8 = {0, 0, 0, 0, 0, 0, 0, 0};
  __builtin_amdgcn_tensor_load_to_lds(g0, g1, z4, z4, z8, 0);
#else
  __builtin_amdgcn_tensor_load_to_lds(g0, g1, z4, z4, 0);
#endif
}

// ---------------------------------------------------------------------------
// Kernel 1: fused 1x1-conv MLP  (x:(B,256,H,W) -> attractors:(pix,16) in ws)
// 256 threads = 8 waves; each wave owns 16 consecutive pixels (M dimension).
// GEMM1: M=16 pix, N=128, K=256   (8 wmma tiles x 8 k-steps, A fed by TDM)
// GEMM2: M=16 pix, N=16,  K=128   (1 wmma tile  x 4 k-steps)
// ---------------------------------------------------------------------------
__global__ __launch_bounds__(256) void mlp_attractors_wmma(
    const float* __restrict__ x,  const float* __restrict__ w1,
    const float* __restrict__ b1, const float* __restrict__ w2,
    const float* __restrict__ b2, float* __restrict__ att)
{
  __shared__ _Float16 w1s[kCmid * W1S];        // 67584 B
  __shared__ _Float16 w2s[kAtt  * W2S];        //  4352 B
  __shared__ _Float16 hs [8 * 16 * HS];        // 34816 B (per-wave h tiles)
  __shared__ float    xs [8 * 2 * XBUF];       // 40960 B (per-wave x dbl-buffer)

  const int tid  = threadIdx.x;
  const int wave = tid >> 5;
  const int lane = tid & 31;
  const int lh   = lane >> 4;   // lane half selects K sub-range (ISA A/B layout)
  const int l16  = lane & 15;

  const int group = blockIdx.x * 8 + wave;     // 16-pixel group id
  const int pix0  = group * 16;
  const int bidx  = pix0 / kHW;                // groups never straddle batches
  const int hw0   = pix0 % kHW;
  const float* gbase = x + (size_t)bidx * kCin * kHW + hw0;  // tile origin

  float* xw = xs + wave * (2 * XBUF);          // this wave's staging buffers

  // Kick off the first DMA before the weight-staging barrier (overlap).
  tdm_load_tile(gbase, (unsigned)(uintptr_t)xw);

  // Stage weights to LDS, converting f32 -> f16 once per block.
  for (int i = tid; i < kCmid * kCin; i += 256) {
    const int n = i >> 8, c = i & 255;
    w1s[n * W1S + c] = (_Float16)w1[i];
  }
  for (int i = tid; i < kAtt * kCmid; i += 256) {
    const int n = i >> 7, c = i & 127;
    w2s[n * W2S + c] = (_Float16)w2[i];
  }
  __syncthreads();

  v8f acc[8] = {};   // h accumulators: 8 tiles of 16x16 f32

#pragma unroll
  for (int ks = 0; ks < 8; ++ks) {
    if (ks < 7) {
      // Recycled buffer: make sure our earlier LDS reads of it drained.
      asm volatile("s_wait_dscnt 0x0" ::: "memory");
      tdm_load_tile(gbase + (size_t)(ks + 1) * 32 * kHW,
                    (unsigned)(uintptr_t)(xw + ((ks + 1) & 1) * XBUF));
      // In-order TDM completion: oldest outstanding (k-step ks) is done.
      __builtin_amdgcn_s_wait_tensorcnt(1);
    } else {
      __builtin_amdgcn_s_wait_tensorcnt(0);
    }
    asm volatile("" ::: "memory");   // don't CSE LDS reads across async write

    const float* xt = xw + (ks & 1) * XBUF;   // [channel][pixel], 20-dw rows
    // A fragment (16-bit A 16x32 layout):
    //   elements 0..7  <-> K = 8*lh + j ; elements 8..15 <-> K = 16 + 8*lh + j
    v16h a;
#pragma unroll
    for (int j = 0; j < 8; ++j) {
      const int cl = lh * 8 + j;
      a[j]     = (_Float16)xt[cl * XROW + l16];          // conflict-free banks
      a[j + 8] = (_Float16)xt[(cl + 16) * XROW + l16];
    }

    const int k32 = ks * 32;
#pragma unroll
    for (int nt = 0; nt < 8; ++nt) {
      // B fragment: row = output channel (N = nt*16 + l16), K = 8*lh + 0..7, 16 + 8*lh + 0..7
      const v16h bf = ld16h(&w1s[(nt * 16 + l16) * W1S + k32 + lh * 8]);
      acc[nt] = __builtin_amdgcn_wmma_f32_16x16x32_f16(
          false, a, false, bf, (short)0, acc[nt], false, false);
    }
  }

  // Epilogue 1: +b1, ReLU, f16, scatter into per-wave LDS h tile [pixel][chan]
  _Float16* hrow = hs + wave * 16 * HS;
#pragma unroll
  for (int nt = 0; nt < 8; ++nt) {
    const int   cc   = nt * 16 + l16;
    const float bias = b1[cc];
#pragma unroll
    for (int r = 0; r < 8; ++r) {
      float v = acc[nt][r] + bias;            // D layout: M = r + 8*lh, N = l16
      v = fmaxf(v, 0.0f);
      hrow[(r + lh * 8) * HS + cc] = (_Float16)v;
    }
  }
  // Same wave produces & consumes hrow; per-wave LDS ops are in-order and the
  // compiler inserts the dscnt waits for the dependent loads below.

  v8f acc2 = {};
#pragma unroll
  for (int ks = 0; ks < 4; ++ks) {
    const int k32 = ks * 32;
    const _Float16* ap = &hrow[l16 * HS + k32 + lh * 8];
    const v8h alo = *(const v8h*)ap;          // K = k32 + 8lh + 0..7
    const v8h ahi = *(const v8h*)(ap + 16);   // K = k32 + 16 + 8lh + 0..7
    const v16h a2 = __builtin_shufflevector(alo, ahi,
        0,1,2,3,4,5,6,7,8,9,10,11,12,13,14,15);
    const v16h bf = ld16h(&w2s[l16 * W2S + k32 + lh * 8]);
    acc2 = __builtin_amdgcn_wmma_f32_16x16x32_f16(
        false, a2, false, bf, (short)0, acc2, false, false);
  }

  // Epilogue 2: +b2, softplus, store pixel-major (lanes 0-15 -> 64B stores)
  const float bias2 = b2[l16];
#pragma unroll
  for (int r = 0; r < 8; ++r) {
    float v  = acc2[r] + bias2;
    float sp = (v > 20.0f) ? v : log1pf(__expf(v));
    att[(size_t)(pix0 + r + lh * 8) * kAtt + l16] = sp;
  }
}

// ---------------------------------------------------------------------------
// Kernel 2: bilinear-upsampled bin centers + inverse-attractor accumulation.
// One thread per pixel; 16 attractors in registers; loop 64 bins.
// ---------------------------------------------------------------------------
__global__ __launch_bounds__(256) void attractor_update(
    const float* __restrict__ att, const float* __restrict__ prev,
    float* __restrict__ out)
{
  const int pix = blockIdx.x * 256 + threadIdx.x;
  const int b   = pix >> 14;          // / kHW
  const int rem = pix & (kHW - 1);
  const int y   = rem >> 7;           // / kW
  const int xc  = rem & 127;

  float a[kAtt];
  const float* ap = att + (size_t)pix * kAtt;   // 64B contiguous per thread
#pragma unroll
  for (int t = 0; t < kAtt; ++t) a[t] = ap[t];

  // align_corners=True resize 64 -> 128
  const float s  = 63.0f / 127.0f;
  const float fy = y * s;
  const int   y0 = (int)fy;
  const float wy = fy - (float)y0;
  const int   y1 = min(y0 + 1, kPH - 1);
  const float fx = xc * s;
  const int   x0 = (int)fx;
  const float wx = fx - (float)x0;
  const int   x1 = min(x0 + 1, kPW - 1);
  const float w00 = (1.0f - wy) * (1.0f - wx), w01 = (1.0f - wy) * wx;
  const float w10 = wy * (1.0f - wx),          w11 = wy * wx;

  const float* pb = prev + (size_t)b * kBins * kPH * kPW;
  for (int j = 0; j < kBins; ++j) {
    const float* pj = pb + (size_t)j * (kPH * kPW);  // L2-resident (8 MB total)
    const float bc = pj[y0 * kPW + x0] * w00 + pj[y0 * kPW + x1] * w01
                   + pj[y1 * kPW + x0] * w10 + pj[y1 * kPW + x1] * w11;
    float d = 0.0f;
#pragma unroll
    for (int t = 0; t < kAtt; ++t) {
      const float dx = a[t] - bc;
      d += dx * __builtin_amdgcn_rcpf(fmaf(300.0f * dx, dx, 1.0f));
    }
    const float  o  = bc + d;
    const size_t oi = ((size_t)(b * kBins + j) * kH + y) * kW + xc;  // coalesced
    out[oi]         = o;   // tuple element 0
    out[oi + kOut1] = o;   // tuple element 1 (duplicate)
  }
}

extern "C" void kernel_launch(void* const* d_in, const int* in_sizes, int n_in,
                              void* d_out, int out_size, void* d_ws, size_t ws_size,
                              hipStream_t stream) {
  const float* x    = (const float*)d_in[0];
  const float* prev = (const float*)d_in[1];
  const float* w1   = (const float*)d_in[2];
  const float* b1   = (const float*)d_in[3];
  const float* w2   = (const float*)d_in[4];
  const float* b2   = (const float*)d_in[5];

  float* att = (float*)d_ws;  // needs kPix * 16 * 4 = 8.4 MB of scratch

  mlp_attractors_wmma<<<kPix / 128, 256, 0, stream>>>(x, w1, b1, w2, b2, att);
  attractor_update  <<<kPix / 256, 256, 0, stream>>>(att, prev, (float*)d_out);
}
